// CrossAttention_29549374997155
// MI455X (gfx1250) — compile-verified, matrix operand order checked by reading, not run on hardware
//
#include <hip/hip_runtime.h>
#include <hip/hip_bf16.h>
#include <math.h>

typedef __attribute__((ext_vector_type(16))) __bf16 v16bf;
typedef __attribute__((ext_vector_type(8)))  __bf16 v8bf;
typedef __attribute__((ext_vector_type(8)))  float  v8f;

constexpr int B  = 256;
constexpr int U  = 128;
constexpr int P  = 64;
constexpr int H  = 768;
constexpr int K2 = U * P;   // 8192 flattened contraction for the conv-as-einsum

// ---------------------------------------------------------------------------
// Kernel 1: inverse L2 norms per row, wave-per-row (wave32).
// src layout [len, B, H]; store inv at [b*len + i] for later [B,len] access.
// Side effect: pulls utt/pheno (150 MB total) into the 192 MB L2 so the
// cosine GEMM's re-read is L2-resident.
// ---------------------------------------------------------------------------
__global__ void norms_kernel(const float* __restrict__ src,
                             float* __restrict__ inv_out, int len) {
  int wave = (blockIdx.x * blockDim.x + threadIdx.x) >> 5;
  int lane = threadIdx.x & 31;
  if (wave >= len * B) return;
  int i = wave / B, b = wave % B;
  const float* row = src + (size_t)wave * H;
  float s = 0.f;
#pragma unroll
  for (int j = 0; j < H; j += 128) {                 // 768 = 6 * 32 lanes * 4
    float4 v = *(const float4*)(row + j + lane * 4);
    s += v.x * v.x + v.y * v.y + v.z * v.z + v.w * v.w;
  }
#pragma unroll
  for (int m = 16; m; m >>= 1) s += __shfl_xor(s, m, 32);
  if (lane == 0) inv_out[b * len + i] = 1.0f / fmaxf(sqrtf(s), 1e-8f);
}

// ---------------------------------------------------------------------------
// Kernel 2: stage weights as bf16. w_utt[o,u,p] copied; w_pheno[o,p,u]
// transposed to [o, u*P+p] so both logits GEMMs contract the same K layout.
// ---------------------------------------------------------------------------
__global__ void wconv_kernel(const float* __restrict__ w_utt,
                             const float* __restrict__ w_pheno,
                             __bf16* __restrict__ wu_bf,
                             __bf16* __restrict__ wp_bf) {
  int idx = blockIdx.x * blockDim.x + threadIdx.x;
  if (idx < U * K2) wu_bf[idx] = (__bf16)w_utt[idx];
  if (idx < P * K2) {
    int o = idx / K2, r = idx % K2;
    int u = r / P, p = r % P;
    wp_bf[idx] = (__bf16)w_pheno[(size_t)o * K2 + p * U + u];
  }
}

// ---------------------------------------------------------------------------
// Fragment builders. ISA 16-bit A-matrix 16x32 layout (wave32):
//   lane<16  -> row=lane,    K = {k0..k0+7} in v[0..7], {k0+16..k0+23} in v[8..15], k0=0
//   lane>=16 -> row=lane-16, same with k0=8
// B (32x16) loaded column-major == same addressing over the "column" rows.
// ---------------------------------------------------------------------------
__device__ __forceinline__ v16bf load_frag_f32(const float* __restrict__ p) {
  float4 a0 = *(const float4*)(p);
  float4 a1 = *(const float4*)(p + 4);
  float4 b0 = *(const float4*)(p + 16);
  float4 b1 = *(const float4*)(p + 20);
  v16bf r;
  r[0]  = (__bf16)a0.x; r[1]  = (__bf16)a0.y;
  r[2]  = (__bf16)a0.z; r[3]  = (__bf16)a0.w;
  r[4]  = (__bf16)a1.x; r[5]  = (__bf16)a1.y;
  r[6]  = (__bf16)a1.z; r[7]  = (__bf16)a1.w;
  r[8]  = (__bf16)b0.x; r[9]  = (__bf16)b0.y;
  r[10] = (__bf16)b0.z; r[11] = (__bf16)b0.w;
  r[12] = (__bf16)b1.x; r[13] = (__bf16)b1.y;
  r[14] = (__bf16)b1.z; r[15] = (__bf16)b1.w;
  return r;
}

__device__ __forceinline__ v16bf load_frag_bf16(const __bf16* __restrict__ p) {
  v8bf lo = *(const v8bf*)(p);        // 16B aligned chunks
  v8bf hi = *(const v8bf*)(p + 16);
  v16bf r;
#pragma unroll
  for (int j = 0; j < 8; ++j) { r[j] = lo[j]; r[j + 8] = hi[j]; }
  return r;
}

// ---------------------------------------------------------------------------
// Kernel 3: per-batch cosine matrix c[b] (128x64) via bf16 WMMA, f32 accum.
// Dot products are computed UNNORMALIZED; the rank-1 normalization
// (inv_nu[m] * inv_np[n]) commutes with the GEMM and is applied in the
// epilogue, keeping the hot K-loop free of per-element scaling VALU.
// One block (8 waves) per batch; wave w owns rows [16w,16w+16) x all 64 cols.
// utt [U,B,H]: u[b,i,:] is a contiguous H-run; pheno rows are B's columns.
// ---------------------------------------------------------------------------
__global__ void cosine_wmma_kernel(const float* __restrict__ utt,
                                   const float* __restrict__ pheno,
                                   const float* __restrict__ inv_nu,
                                   const float* __restrict__ inv_np,
                                   __bf16* __restrict__ c_bf) {
  int b    = blockIdx.x;
  int wave = threadIdx.x >> 5;
  int lane = threadIdx.x & 31;
  int rt   = lane & 15;
  int hi   = lane >> 4;
  int k0   = hi ? 8 : 0;

  int u_row = wave * 16 + rt;
  const float* Arow = utt + ((size_t)u_row * B + b) * H + k0;

  const float* Brow[4];
#pragma unroll
  for (int nt = 0; nt < 4; ++nt) {
    int p_col = nt * 16 + rt;
    Brow[nt]  = pheno + ((size_t)p_col * B + b) * H + k0;
  }

  v8f acc[4] = {};
  for (int kk = 0; kk < H; kk += 32) {
    v16bf a = load_frag_f32(Arow + kk);
#pragma unroll
    for (int nt = 0; nt < 4; ++nt) {
      v16bf bb = load_frag_f32(Brow[nt] + kk);
      acc[nt] = __builtin_amdgcn_wmma_f32_16x16x32_bf16(
          false, a, false, bb, (short)0, acc[nt], false, false);
    }
  }

  // Epilogue: D layout has VGPR r -> (M = r + 8*hi, N = rt).
  // Row scales for this lane's 8 M values are 8 consecutive floats.
  const float* su_base = inv_nu + b * U + wave * 16 + 8 * hi;
  float4 su0 = *(const float4*)(su_base);
  float4 su1 = *(const float4*)(su_base + 4);
  float su[8] = {su0.x, su0.y, su0.z, su0.w, su1.x, su1.y, su1.z, su1.w};

  __bf16* cb = c_bf + (size_t)b * K2;
#pragma unroll
  for (int nt = 0; nt < 4; ++nt) {
    float sp = inv_np[b * P + nt * 16 + rt];
#pragma unroll
    for (int r = 0; r < 8; ++r) {
      int m = r + 8 * hi;
      cb[(wave * 16 + m) * P + nt * 16 + rt] = (__bf16)(acc[nt][r] * su[r] * sp);
    }
  }
}

// ---------------------------------------------------------------------------
// Kernel 4: logits GEMM. A = c_bf [B, K2] row-major, W = [N, K2] row-major
// (== column-major B-matrix). 4 waves/block, each owns one 16x16 tile.
// Epilogue adds bias and writes f32 logits straight into d_out.
// ---------------------------------------------------------------------------
__global__ void logits_wmma_kernel(const __bf16* __restrict__ c_bf,
                                   const __bf16* __restrict__ W,
                                   const float* __restrict__ bias,
                                   float* __restrict__ out, int N) {
  int mtile = blockIdx.x;
  int wave  = threadIdx.x >> 5;
  int ntile = blockIdx.y * 4 + wave;
  int lane  = threadIdx.x & 31;
  int rt    = lane & 15;
  int hi    = lane >> 4;
  int k0    = hi ? 8 : 0;

  const __bf16* Arow = c_bf + (size_t)(mtile * 16 + rt) * K2 + k0;
  const __bf16* Wrow = W    + (size_t)(ntile * 16 + rt) * K2 + k0;

  v8f acc = {};
  for (int kk = 0; kk < K2; kk += 32) {
    v16bf a = load_frag_bf16(Arow + kk);
    v16bf w = load_frag_bf16(Wrow + kk);
    acc = __builtin_amdgcn_wmma_f32_16x16x32_bf16(
        false, a, false, w, (short)0, acc, false, false);
  }

  int n = ntile * 16 + rt;
  float bv = bias[n];
#pragma unroll
  for (int r = 0; r < 8; ++r) {
    int m = r + 8 * hi;
    out[(size_t)(mtile * 16 + m) * N + n] = acc[r] + bv;
  }
}

// ---------------------------------------------------------------------------
// Kernel 5: in-place row softmax, wave-per-row (n = 128 or 64).
// ---------------------------------------------------------------------------
__global__ void softmax_kernel(float* __restrict__ x, int n) {
  int row  = (blockIdx.x * blockDim.x + threadIdx.x) >> 5;
  int lane = threadIdx.x & 31;
  if (row >= B) return;
  float* p = x + (size_t)row * n;
  int per = n >> 5;                                  // 4 or 2
  float vals[4];
  float mx = -1e30f;
  for (int j = 0; j < per; ++j) { vals[j] = p[lane + 32 * j]; mx = fmaxf(mx, vals[j]); }
#pragma unroll
  for (int m = 16; m; m >>= 1) mx = fmaxf(mx, __shfl_xor(mx, m, 32));
  float s = 0.f;
  for (int j = 0; j < per; ++j) { vals[j] = __expf(vals[j] - mx); s += vals[j]; }
#pragma unroll
  for (int m = 16; m; m >>= 1) s += __shfl_xor(s, m, 32);
  float inv = 1.0f / s;
  for (int j = 0; j < per; ++j) p[lane + 32 * j] = vals[j] * inv;
}

// ---------------------------------------------------------------------------
extern "C" void kernel_launch(void* const* d_in, const int* in_sizes, int n_in,
                              void* d_out, int out_size, void* d_ws, size_t ws_size,
                              hipStream_t stream) {
  const float* utt     = (const float*)d_in[0];
  const float* pheno   = (const float*)d_in[1];
  const float* w_utt   = (const float*)d_in[2];
  const float* b_utt   = (const float*)d_in[3];
  const float* w_pheno = (const float*)d_in[4];
  const float* b_pheno = (const float*)d_in[5];
  (void)in_sizes; (void)n_in; (void)out_size; (void)ws_size;

  char* ws = (char*)d_ws;
  size_t off = 0;
  auto alloc = [&](size_t bytes) {
    void* p = ws + off;
    off = (off + bytes + 255) & ~(size_t)255;
    return p;
  };
  float*  inv_nu = (float*)alloc((size_t)B * U * sizeof(float));
  float*  inv_np = (float*)alloc((size_t)B * P * sizeof(float));
  __bf16* wu_bf  = (__bf16*)alloc((size_t)U * K2 * sizeof(__bf16));
  __bf16* wp_bf  = (__bf16*)alloc((size_t)P * K2 * sizeof(__bf16));
  __bf16* c_bf   = (__bf16*)alloc((size_t)B * K2 * sizeof(__bf16));

  float* row_logits = (float*)d_out;            // [B, U]
  float* col_logits = (float*)d_out + B * U;    // [B, P]

  norms_kernel<<<(B * U) / 8, 256, 0, stream>>>(utt, inv_nu, U);
  norms_kernel<<<(B * P) / 8, 256, 0, stream>>>(pheno, inv_np, P);
  wconv_kernel<<<(U * K2) / 256, 256, 0, stream>>>(w_utt, w_pheno, wu_bf, wp_bf);

  cosine_wmma_kernel<<<B, 256, 0, stream>>>(utt, pheno, inv_nu, inv_np, c_bf);

  logits_wmma_kernel<<<dim3(B / 16, U / 64), 128, 0, stream>>>(c_bf, wu_bf, b_utt, row_logits, U);
  logits_wmma_kernel<<<dim3(B / 16, P / 64), 128, 0, stream>>>(c_bf, wp_bf, b_pheno, col_logits, P);

  softmax_kernel<<<B / 8, 256, 0, stream>>>(row_logits, U);
  softmax_kernel<<<B / 8, 256, 0, stream>>>(col_logits, P);
}
